// ASTGCN_26560077758760
// MI455X (gfx1250) — matrix operand order repaired and checked by reading, not compiled, not run on hardware
//
#include <hip/hip_runtime.h>
#include <hip/hip_bf16.h>

typedef __attribute__((ext_vector_type(16))) _Float16 v16h;
typedef __attribute__((ext_vector_type(8)))  _Float16 v8h;
typedef __attribute__((ext_vector_type(8)))  float    v8f;

// Flip to 0 if the assembler rejects the async-to-LDS mnemonics.
#ifndef ASTGCN_USE_ASYNC
#define ASTGCN_USE_ASYNC 1
#endif

namespace {
constexpr int Bb   = 16;    // batch
constexpr int Nn   = 1024;  // nodes
constexpr int Tt   = 24;    // time steps
constexpr int Kch  = 3;     // Chebyshev order
constexpr int Fch  = 64;    // F_CHEB
constexpr int Ftm  = 64;    // F_TIME
constexpr int OUTD = 12;
}

__device__ __forceinline__ float sigm(float v) { return 1.0f / (1.0f + expf(-v)); }

// Wave-relative LDS byte offset: flat shared-aperture addresses carry the LDS
// offset in addr[31:0] (CDNA5 ISA 10.2 aperture mapping).
__device__ __forceinline__ unsigned lds_off(const void* p) {
  return (unsigned)(unsigned long long)p;
}

// ---------------------------------------------------------------------------
// Temporal attention
// ---------------------------------------------------------------------------
__global__ void k_xu1(const float* __restrict__ x, const float* __restrict__ U1,
                      float* __restrict__ o, int C, int total) {
  int i = blockIdx.x * 256 + threadIdx.x;
  if (i >= total) return;
  int c = i % C, t = (i / C) % Tt, b = i / (C * Tt);
  const float* xb = x + ((size_t)b * Nn * C + c) * Tt + t;
  float s = 0.f;
  for (int n = 0; n < Nn; ++n) s += xb[(size_t)n * C * Tt] * U1[n];
  o[i] = s;
}

__global__ void k_lhs(const float* __restrict__ xu1, const float* __restrict__ U2,
                      float* __restrict__ o, int C, int total) {
  int i = blockIdx.x * 256 + threadIdx.x;
  if (i >= total) return;
  int n = i % Nn, t = (i / Nn) % Tt, b = i / (Nn * Tt);
  float s = 0.f;
  for (int c = 0; c < C; ++c) s += xu1[(b * Tt + t) * C + c] * U2[c * Nn + n];
  o[i] = s;
}

__global__ void k_rhs(const float* __restrict__ x, const float* __restrict__ U3,
                      float* __restrict__ o, int C, int total) {
  int i = blockIdx.x * 256 + threadIdx.x;
  if (i >= total) return;
  int t = i % Tt, n = (i / Tt) % Nn, b = i / (Tt * Nn);
  float s = 0.f;
  for (int c = 0; c < C; ++c) s += U3[c] * x[((size_t)(b * Nn + n) * C + c) * Tt + t];
  o[i] = s;
}

__global__ void k_prod_t(const float* __restrict__ lhs, const float* __restrict__ rhs,
                         float* __restrict__ o, int total) {
  int i = blockIdx.x * 256 + threadIdx.x;
  if (i >= total) return;
  int ss = i % Tt, t = (i / Tt) % Tt, b = i / (Tt * Tt);
  float s = 0.f;
  for (int n = 0; n < Nn; ++n)
    s += lhs[((size_t)b * Tt + t) * Nn + n] * rhs[((size_t)b * Nn + n) * Tt + ss];
  o[i] = s;
}

__global__ void k_E(const float* __restrict__ prod, const float* __restrict__ be,
                    const float* __restrict__ Ve, float* __restrict__ E, int total) {
  int i = blockIdx.x * 256 + threadIdx.x;
  if (i >= total) return;
  int ss = i % Tt, b = i / Tt;
  float tmp[Tt];
  for (int t = 0; t < Tt; ++t) {
    float a = 0.f;
    for (int u = 0; u < Tt; ++u) {
      float pv = prod[(b * Tt + u) * Tt + ss] + be[u * Tt + ss];
      a += Ve[t * Tt + u] * sigm(pv);
    }
    tmp[t] = a;
  }
  float m = tmp[0];
  for (int t = 1; t < Tt; ++t) m = fmaxf(m, tmp[t]);
  float den = 0.f;
  for (int t = 0; t < Tt; ++t) { tmp[t] = expf(tmp[t] - m); den += tmp[t]; }
  float inv = 1.f / den;
  for (int t = 0; t < Tt; ++t) E[(b * Tt + t) * Tt + ss] = tmp[t] * inv;
}

__global__ void k_xtat(const float* __restrict__ x, const float* __restrict__ E,
                       float* __restrict__ o, int C, int total) {
  int i = blockIdx.x * 256 + threadIdx.x;
  if (i >= total) return;
  int ss = i % Tt;
  size_t row = (size_t)(i / Tt);            // (b*Nn+n)*C + c
  int b = i / (Tt * C * Nn);
  const float* xb = x + row * Tt;
  float s = 0.f;
  for (int t = 0; t < Tt; ++t) s += xb[t] * E[(b * Tt + t) * Tt + ss];
  o[i] = s;
}

// ---------------------------------------------------------------------------
// Spatial attention (staging for WMMA GEMM; all GEMM B-operands are stored
// TRANSPOSED, i.e. (Nt x K) row-major, so GEMM tiles stage identically to A)
// ---------------------------------------------------------------------------
__global__ void k_xw1(const float* __restrict__ xtat, const float* __restrict__ W1,
                      float* __restrict__ o, int total) {
  int i = blockIdx.x * 256 + threadIdx.x;
  if (i >= total) return;
  const float* xb = xtat + (size_t)i * Tt;
  float s = 0.f;
  for (int t = 0; t < Tt; ++t) s += xb[t] * W1[t];
  o[i] = s;
}

__global__ void k_lhsS(const float* __restrict__ xw1, const float* __restrict__ W2,
                       _Float16* __restrict__ o, int C, int total) {
  int i = blockIdx.x * 256 + threadIdx.x;
  if (i >= total) return;
  int kk = i % 32;
  size_t r = (size_t)(i / 32);              // b*Nn + n
  float s = 0.f;
  if (kk < Tt)
    for (int c = 0; c < C; ++c) s += xw1[r * C + c] * W2[c * Tt + kk];
  o[i] = (_Float16)s;
}

// rhs_s in natural layout (B, Nn, 32): Bt[m][kk] = sum_c W3[c]*x_TAt[b,m,c,kk]
__global__ void k_rhsT(const float* __restrict__ xtat, const float* __restrict__ W3,
                       _Float16* __restrict__ o, int C, int total) {
  int i = blockIdx.x * 256 + threadIdx.x;
  if (i >= total) return;
  int kk = i % 32;
  size_t r = (size_t)(i / 32);              // b*Nn + m
  float s = 0.f;
  if (kk < Tt)
    for (int c = 0; c < C; ++c) s += W3[c] * xtat[(r * C + c) * Tt + kk];
  o[i] = (_Float16)s;
}

// S^T tile transpose: o[b,k,m] = sigmoid(P[b,m,k] + bs[m,k]) as f16.
// Grid (Nn/32, Nn/32, Bb), block (32, 8).
__global__ __launch_bounds__(256) void k_sigSt(
    const float* __restrict__ P, const float* __restrict__ bs,
    _Float16* __restrict__ o) {
  __shared__ float tile[32][33];
  const int b  = blockIdx.z;
  const int m0 = blockIdx.y * 32;
  const int k0 = blockIdx.x * 32;
  const int tx = threadIdx.x, ty = threadIdx.y;
  const float* Pb = P + (size_t)b * Nn * Nn;
#pragma unroll
  for (int j = 0; j < 4; ++j) {
    int m = m0 + ty + j * 8;
    tile[ty + j * 8][tx] = sigm(Pb[(size_t)m * Nn + k0 + tx] +
                                bs[(size_t)m * Nn + k0 + tx]);
  }
  __syncthreads();
  _Float16* ob = o + (size_t)b * Nn * Nn;
#pragma unroll
  for (int j = 0; j < 4; ++j) {
    int k = k0 + ty + j * 8;
    ob[(size_t)k * Nn + m0 + tx] = (_Float16)tile[tx][ty + j * 8];
  }
}

__global__ void k_f32tof16(const float* __restrict__ src, _Float16* __restrict__ dst,
                           int total) {
  int i = blockIdx.x * 256 + threadIdx.x;
  if (i >= total) return;
  dst[i] = (_Float16)src[i];
}

__global__ void k_softmax_col(const float* __restrict__ Q, float* __restrict__ sA,
                              int total) {
  int i = blockIdx.x * 256 + threadIdx.x;
  if (i >= total) return;
  int kcol = i % Nn, b = i / Nn;
  size_t base = (size_t)b * Nn * Nn + kcol;
  float m = -1e30f;
  for (int n = 0; n < Nn; ++n) m = fmaxf(m, Q[base + (size_t)n * Nn]);
  float den = 0.f;
  for (int n = 0; n < Nn; ++n) den += expf(Q[base + (size_t)n * Nn] - m);
  float inv = 1.f / den;
  for (int n = 0; n < Nn; ++n)
    sA[base + (size_t)n * Nn] = expf(Q[base + (size_t)n * Nn] - m) * inv;
}

// ---------------------------------------------------------------------------
// Chebyshev staging
// ---------------------------------------------------------------------------
// X^T tile transpose: o[b,j,m] = x_flat[b,m,j] (zero-padded j >= C*Tt), f16.
// Grid (ctpad/32, Nn/32, Bb), block (32, 8).
__global__ __launch_bounds__(256) void k_xft(
    const float* __restrict__ x, _Float16* __restrict__ o, int C, int ctpad) {
  __shared__ float tile[32][33];
  const int b  = blockIdx.z;
  const int n0 = blockIdx.y * 32;
  const int j0 = blockIdx.x * 32;
  const int tx = threadIdx.x, ty = threadIdx.y;
  const int ct = C * Tt;
#pragma unroll
  for (int jj = 0; jj < 4; ++jj) {
    int n = n0 + ty + jj * 8;
    int j = j0 + tx;
    tile[ty + jj * 8][tx] =
        (j < ct) ? x[((size_t)b * Nn + n) * ct + j] : 0.f;
  }
  __syncthreads();
#pragma unroll
  for (int jj = 0; jj < 4; ++jj) {
    int j = j0 + ty + jj * 8;
    o[((size_t)b * ctpad + j) * Nn + n0 + tx] = (_Float16)tile[tx][ty + jj * 8];
  }
}

// AkT[b,n,m] = cheb[k][m,n] * sA[b,m,n]  -- LDS tile transpose, both sides
// coalesced.  Grid (Nn/32, Nn/32, Bb), block (32, 8).
__global__ __launch_bounds__(256) void k_akt(
    const float* __restrict__ chebk, const float* __restrict__ sA,
    _Float16* __restrict__ o) {
  __shared__ float tile[32][33];
  const int b  = blockIdx.z;
  const int m0 = blockIdx.y * 32;           // source row block (m)
  const int n0 = blockIdx.x * 32;           // source col block (n)
  const int tx = threadIdx.x;               // 0..31
  const int ty = threadIdx.y;               // 0..7
  const float* sAb = sA + (size_t)b * Nn * Nn;
#pragma unroll
  for (int j = 0; j < 4; ++j) {
    int m = m0 + ty + j * 8;
    tile[ty + j * 8][tx] = chebk[(size_t)m * Nn + n0 + tx] *
                           sAb[(size_t)m * Nn + n0 + tx];
  }
  __syncthreads();
  _Float16* ob = o + (size_t)b * Nn * Nn;
#pragma unroll
  for (int j = 0; j < 4; ++j) {
    int n = n0 + ty + j * 8;
    ob[(size_t)n * Nn + m0 + tx] = (_Float16)tile[tx][ty + j * 8];
  }
}

__global__ void k_theta(const float* __restrict__ Y, const float* __restrict__ Th,
                        float* __restrict__ G, int C, int ctpad, int accum, int total) {
  int i = blockIdx.x * 256 + threadIdx.x;
  if (i >= total) return;
  int t = i % Tt, o_ = (i / Tt) % Fch, n = (i / (Tt * Fch)) % Nn, b = i / (Tt * Fch * Nn);
  const float* yb = Y + (size_t)(b * Nn + n) * ctpad + t;
  float s = 0.f;
  for (int c = 0; c < C; ++c) s += yb[c * Tt] * Th[c * Fch + o_];
  if (accum) G[i] += s; else G[i] = s;
}

// ---------------------------------------------------------------------------
// Fused residual conv + temporal conv, then LayerNorm
// ---------------------------------------------------------------------------
__global__ void k_convs(const float* __restrict__ x, const float* __restrict__ G,
                        const float* __restrict__ tcw, const float* __restrict__ tcb,
                        const float* __restrict__ rcw, const float* __restrict__ rcb,
                        float* __restrict__ Yb, int C, int total) {
  int i = blockIdx.x * 256 + threadIdx.x;
  if (i >= total) return;
  int t = i % Tt, f = (i / Tt) % Ftm, n = (i / (Tt * Ftm)) % Nn, b = i / (Tt * Ftm * Nn);
  float acc = rcb[f] + tcb[f];
  const float* xb = x + (size_t)(b * Nn + n) * C * Tt + t;
  for (int c = 0; c < C; ++c) acc += rcw[f * C + c] * xb[c * Tt];
  const float* gb = G + (size_t)(b * Nn + n) * Fch * Tt;
  for (int c2 = 0; c2 < Fch; ++c2) {
    const float* gr = gb + c2 * Tt;
    const float* wv = tcw + (f * Fch + c2) * 3;
    if (t > 0)      acc += wv[0] * fmaxf(gr[t - 1], 0.f);
                    acc += wv[1] * fmaxf(gr[t],     0.f);
    if (t < Tt - 1) acc += wv[2] * fmaxf(gr[t + 1], 0.f);
  }
  Yb[i] = fmaxf(acc, 0.f);
}

__global__ void k_ln(const float* __restrict__ Yb, const float* __restrict__ g,
                     const float* __restrict__ bt, float* __restrict__ H, int total) {
  int i = blockIdx.x * 256 + threadIdx.x;
  if (i >= total) return;
  int t = i % Tt, n = (i / Tt) % Nn, b = i / (Tt * Nn);
  size_t base = (size_t)(b * Nn + n) * Ftm * Tt + t;
  float mu = 0.f;
  for (int f = 0; f < Ftm; ++f) mu += Yb[base + f * Tt];
  mu *= (1.f / Ftm);
  float var = 0.f;
  for (int f = 0; f < Ftm; ++f) { float d = Yb[base + f * Tt] - mu; var += d * d; }
  var *= (1.f / Ftm);
  float inv = rsqrtf(var + 1e-5f);
  for (int f = 0; f < Ftm; ++f)
    H[base + f * Tt] = (Yb[base + f * Tt] - mu) * inv * g[f] + bt[f];
}

__global__ void k_final(const float* __restrict__ H, const float* __restrict__ fw,
                        const float* __restrict__ fb, float* __restrict__ out, int total) {
  int i = blockIdx.x * 256 + threadIdx.x;
  if (i >= total) return;
  int o = i % OUTD, n = (i / OUTD) % Nn, b = i / (OUTD * Nn);
  const float* hb = H + (size_t)(b * Nn + n) * Ftm * Tt;
  float acc = fb[o];
  for (int t = 0; t < Tt; ++t)
    for (int f = 0; f < Ftm; ++f)
      acc += fw[(o * Tt + t) * Ftm + f] * hb[f * Tt + t];
  out[i] = acc;
}

// ---------------------------------------------------------------------------
// Tiled batched GEMM: C[b](MxNt) = A[b](MxK) * Bt[b](NtxK)^T, f16 in, f32 out.
// Both operands are (rows x K) row-major, so A and B tiles stage identically:
// 128 rows x 32 halves, 2 threads/row, two async b128s per thread straight
// into LDS (ASYNCcnt-tracked).  Double-buffered: tile s+1 is in flight while
// tile s feeds 8 WMMAs/wave.  One barrier per iteration.
// Requires M % 128 == 0, Nt % 128 == 0, K % 32 == 0 (guaranteed by callers).
// ---------------------------------------------------------------------------
__global__ __launch_bounds__(256) void k_gemm_f16(
    const _Float16* __restrict__ A, unsigned long long strA,
    const _Float16* __restrict__ Bt, unsigned long long strB,
    float* __restrict__ C, unsigned long long strC,
    int M, int Kdim, int Nt) {
  __shared__ _Float16 As[2][128][32];
  __shared__ _Float16 Bs[2][128][32];

  const int b    = blockIdx.z;
  const int mB   = blockIdx.y * 128;
  const int nB   = blockIdx.x * 128;
  const int tid  = threadIdx.x;
  const int lane = tid & 31;
  const int w    = tid >> 5;
  const int wm   = w >> 1;                 // 0..3 : 32-row strip
  const int wn   = w & 1;                  // 0..1 : 64-col strip
  const int l16  = lane & 15;
  const int hi   = lane >> 4;              // 0 for lanes 0-15, 1 for 16-31

  const _Float16* Ab  = A  + (size_t)b * strA;
  const _Float16* Btb = Bt + (size_t)b * strB;
  float* Cb = C + (size_t)b * strC;

  // Tile staging: thread -> (row = tid>>1, col = (tid&1)*16), 32 bytes.
  const int arow = tid >> 1;
  const int acol = (tid & 1) * 16;

  auto stageTo = [&](const _Float16* src, _Float16* dst) {
#if ASTGCN_USE_ASYNC
    unsigned l0 = lds_off(dst);
    unsigned long long g0 = (unsigned long long)src;
    asm volatile("global_load_async_to_lds_b128 %0, %1, off"
                 :: "v"(l0), "v"(g0) : "memory");
    asm volatile("global_load_async_to_lds_b128 %0, %1, off offset:16"
                 :: "v"(l0), "v"(g0) : "memory");
#else
    *(v8h*)dst       = *(const v8h*)(src);
    *(v8h*)(dst + 8) = *(const v8h*)(src + 8);
#endif
  };

  v8f zero = {0.f, 0.f, 0.f, 0.f, 0.f, 0.f, 0.f, 0.f};
  v8f acc[2][4];
#pragma unroll
  for (int mi = 0; mi < 2; ++mi)
#pragma unroll
    for (int ni = 0; ni < 4; ++ni) acc[mi][ni] = zero;

  const int nsteps = Kdim / 32;
  stageTo(Ab  + (size_t)(mB + arow) * Kdim + acol, &As[0][arow][acol]);
  stageTo(Btb + (size_t)(nB + arow) * Kdim + acol, &Bs[0][arow][acol]);

  for (int s = 0; s < nsteps; ++s) {
    const int cur = s & 1;
#if ASTGCN_USE_ASYNC
    asm volatile("s_wait_asynccnt 0x0" ::: "memory");
#endif
    __syncthreads();                      // tile s ready; buffer cur^1 free

    if (s + 1 < nsteps) {                 // launch tile s+1 into other buffer
      int kn = (s + 1) * 32;
      stageTo(Ab  + (size_t)(mB + arow) * Kdim + kn + acol, &As[cur ^ 1][arow][acol]);
      stageTo(Btb + (size_t)(nB + arow) * Kdim + kn + acol, &Bs[cur ^ 1][arow][acol]);
    }
    if (s + 2 < nsteps) {                 // CDNA5 global_prefetch_b8, depth 2
      __builtin_prefetch(Ab  + (size_t)(mB + arow) * Kdim + (s + 2) * 32 + acol, 0, 0);
      __builtin_prefetch(Btb + (size_t)(nB + arow) * Kdim + (s + 2) * 32 + acol, 0, 0);
    }

    // ---- fragments per documented CDNA5 VGPR layouts ----
    v16h afr[2], bfr[4];
#pragma unroll
    for (int mi = 0; mi < 2; ++mi) {
      int rm = wm * 32 + mi * 16 + l16;
      int lo = hi * 8;                     // lanes<16: K 0..7 & 16..23; lanes>=16: +8
      v8h a0 = *(const v8h*)&As[cur][rm][lo];
      v8h a1 = *(const v8h*)&As[cur][rm][16 + lo];
      afr[mi] = __builtin_shufflevector(a0, a1, 0, 1, 2, 3, 4, 5, 6, 7,
                                        8, 9, 10, 11, 12, 13, 14, 15);
    }
#pragma unroll
    for (int ni = 0; ni < 4; ++ni) {
      int rn = wn * 64 + ni * 16 + l16;
      int cb = hi * 8;                     // lanes<16: K 0..7 & 16..23; lanes>=16: +8
      v8h b0 = *(const v8h*)&Bs[cur][rn][cb];
      v8h b1 = *(const v8h*)&Bs[cur][rn][cb + 16];
      bfr[ni] = __builtin_shufflevector(b0, b1, 0, 1, 2, 3, 4, 5, 6, 7,
                                        8, 9, 10, 11, 12, 13, 14, 15);
    }
#pragma unroll
    for (int mi = 0; mi < 2; ++mi)
#pragma unroll
      for (int ni = 0; ni < 4; ++ni)
        acc[mi][ni] = __builtin_amdgcn_wmma_f32_16x16x32_f16(
            false, afr[mi], false, bfr[ni], (short)0, acc[mi][ni], false, false);
  }

  // Store: VGPR r -> row r (lanes 0-15) or 8+r (lanes 16-31); col = lane&15.
#pragma unroll
  for (int mi = 0; mi < 2; ++mi) {
    int mBase = mB + wm * 32 + mi * 16 + hi * 8;
#pragma unroll
    for (int ni = 0; ni < 4; ++ni) {
      int gn = nB + wn * 64 + ni * 16 + l16;
#pragma unroll
      for (int r = 0; r < 8; ++r)
        Cb[(size_t)(mBase + r) * Nt + gn] = acc[mi][ni][r];
    }
  }
}

// ---------------------------------------------------------------------------
// Host orchestration
// ---------------------------------------------------------------------------
namespace {

struct WS {
  float *XTAT, *P, *Q, *Y, *G, *H0;
  float *XU1, *LHS, *RHS, *PROD, *E, *XW1;
  _Float16 *S16, *XF16, *VS16, *LHSS, *RHSST;
};

inline dim3 gsz(long long n) { return dim3((unsigned)((n + 255) / 256)); }

inline void gemm(hipStream_t st, const _Float16* A, unsigned long long sA,
                 const _Float16* Bt, unsigned long long sB,
                 float* C, unsigned long long sC, int M, int K, int Nt) {
  dim3 g(Nt / 128, M / 128, Bb);
  k_gemm_f16<<<g, 256, 0, st>>>(A, sA, Bt, sB, C, sC, M, K, Nt);
}

// p[] order (insertion order of _make_block_params):
// 0:U1 1:U2 2:U3 3:be 4:Ve 5:W1 6:W2 7:W3 8:bs 9:Vs 10:Theta
// 11:tc_w 12:tc_b 13:rc_w 14:rc_b 15:ln_g 16:ln_b
void run_block(hipStream_t st, const float* x, int C, int ctpad,
               const float* const* p, const float* cheb, const WS& w,
               float* Hout) {
  long long n;
  // ---- temporal attention ----
  n = (long long)Bb * Tt * C;
  k_xu1<<<gsz(n), 256, 0, st>>>(x, p[0], w.XU1, C, (int)n);
  n = (long long)Bb * Tt * Nn;
  k_lhs<<<gsz(n), 256, 0, st>>>(w.XU1, p[1], w.LHS, C, (int)n);
  n = (long long)Bb * Nn * Tt;
  k_rhs<<<gsz(n), 256, 0, st>>>(x, p[2], w.RHS, C, (int)n);
  n = (long long)Bb * Tt * Tt;
  k_prod_t<<<gsz(n), 256, 0, st>>>(w.LHS, w.RHS, w.PROD, (int)n);
  n = (long long)Bb * Tt;
  k_E<<<gsz(n), 256, 0, st>>>(w.PROD, p[3], p[4], w.E, (int)n);
  n = (long long)Bb * Nn * C * Tt;
  k_xtat<<<gsz(n), 256, 0, st>>>(x, w.E, w.XTAT, C, (int)n);
  // ---- spatial attention ----
  n = (long long)Bb * Nn * C;
  k_xw1<<<gsz(n), 256, 0, st>>>(w.XTAT, p[5], w.XW1, (int)n);
  n = (long long)Bb * Nn * 32;
  k_lhsS<<<gsz(n), 256, 0, st>>>(w.XW1, p[6], w.LHSS, C, (int)n);
  n = (long long)Bb * Nn * 32;
  k_rhsT<<<gsz(n), 256, 0, st>>>(w.XTAT, p[7], w.RHSST, C, (int)n);
  // prod_s = lhs_s @ rhs_s^T  (K padded 24->32 with zeros; Bt natural layout)
  gemm(st, w.LHSS, (unsigned long long)Nn * 32, w.RHSST,
       (unsigned long long)Nn * 32, w.P, (unsigned long long)Nn * Nn,
       Nn, 32, Nn);
  // S^T = sigmoid(prod + bs), tiled transpose to (B, k, m) f16
  {
    dim3 tg(Nn / 32, Nn / 32, Bb);
    k_sigSt<<<tg, dim3(32, 8), 0, st>>>(w.P, p[8], w.S16);
  }
  n = (long long)Nn * Nn;
  k_f32tof16<<<gsz(n), 256, 0, st>>>(p[9], w.VS16, (int)n);
  // Q = Vs @ S  (Vs broadcast via stride 0; Bt = S^T)
  gemm(st, w.VS16, 0ULL, w.S16, (unsigned long long)Nn * Nn, w.Q,
       (unsigned long long)Nn * Nn, Nn, Nn, Nn);
  n = (long long)Bb * Nn;
  k_softmax_col<<<gsz(n), 256, 0, st>>>(w.Q, w.P, (int)n);   // sA -> w.P
  // ---- Chebyshev graph conv ----
  {
    dim3 tg(ctpad / 32, Nn / 32, Bb);
    k_xft<<<tg, dim3(32, 8), 0, st>>>(x, w.XF16, C, ctpad);  // X^T (B,ctpad,Nn)
  }
  for (int k = 0; k < Kch; ++k) {
    dim3 tg(Nn / 32, Nn / 32, Bb);
    k_akt<<<tg, dim3(32, 8), 0, st>>>(cheb + (size_t)k * Nn * Nn, w.P, w.S16);
    gemm(st, w.S16, (unsigned long long)Nn * Nn, w.XF16,
         (unsigned long long)Nn * ctpad, w.Y, (unsigned long long)Nn * ctpad,
         Nn, Nn, ctpad);
    n = (long long)Bb * Nn * Fch * Tt;
    k_theta<<<gsz(n), 256, 0, st>>>(w.Y, p[10] + (size_t)k * C * Fch, w.G, C,
                                    ctpad, (k > 0) ? 1 : 0, (int)n);
  }
  // ---- temporal conv + residual (fused), then LayerNorm ----
  float* YB = w.Y;  // Y is dead; alias (equal size: Bb*Nn*1536*4 == Bb*Nn*64*24*4)
  n = (long long)Bb * Nn * Ftm * Tt;
  k_convs<<<gsz(n), 256, 0, st>>>(x, w.G, p[11], p[12], p[13], p[14], YB, C, (int)n);
  n = (long long)Bb * Nn * Tt;
  k_ln<<<gsz(n), 256, 0, st>>>(YB, p[15], p[16], Hout, (int)n);
}

} // namespace

extern "C" void kernel_launch(void* const* d_in, const int* in_sizes, int n_in,
                              void* d_out, int out_size, void* d_ws, size_t ws_size,
                              hipStream_t stream) {
  (void)in_sizes; (void)n_in; (void)out_size; (void)ws_size;

  const float* x0   = (const float*)d_in[0];
  const float* cheb = (const float*)d_in[1];
  const float* P0[17];
  const float* P1[17];
  for (int i = 0; i < 17; ++i) P0[i] = (const float*)d_in[2 + i];
  for (int i = 0; i < 17; ++i) P1[i] = (const float*)d_in[19 + i];
  const float* fw = (const float*)d_in[36];
  const float* fb = (const float*)d_in[37];
  float* out = (float*)d_out;

  // ---- workspace carve (deterministic bump allocator) ----
  char* wp = (char*)d_ws;
  auto take = [&](size_t bytes) -> char* {
    char* p = wp;
    wp += (bytes + 255) & ~(size_t)255;
    return p;
  };
  const size_t NN2 = (size_t)Nn * Nn;
  const size_t BNN = (size_t)Bb * NN2;
  WS w;
  w.XTAT  = (float*)take((size_t)Bb * Nn * Ftm * Tt * 4);   // also block-1 output H1
  w.P     = (float*)take(BNN * 4);
  w.Q     = (float*)take(BNN * 4);
  w.Y     = (float*)take((size_t)Bb * Nn * 1536 * 4);       // also YB (same size)
  w.G     = (float*)take((size_t)Bb * Nn * Fch * Tt * 4);
  w.H0    = (float*)take((size_t)Bb * Nn * Ftm * Tt * 4);
  w.S16   = (_Float16*)take(BNN * 2);                       // S^T, then AkT
  w.XF16  = (_Float16*)take((size_t)Bb * Nn * 1536 * 2);    // X^T
  w.VS16  = (_Float16*)take(NN2 * 2);
  w.LHSS  = (_Float16*)take((size_t)Bb * Nn * 32 * 2);
  w.RHSST = (_Float16*)take((size_t)Bb * Nn * 32 * 2);
  w.XU1   = (float*)take((size_t)Bb * Tt * Ftm * 4);
  w.LHS   = (float*)take((size_t)Bb * Tt * Nn * 4);
  w.RHS   = (float*)take((size_t)Bb * Nn * Tt * 4);
  w.PROD  = (float*)take((size_t)Bb * Tt * Tt * 4);
  w.E     = (float*)take((size_t)Bb * Tt * Tt * 4);
  w.XW1   = (float*)take((size_t)Bb * Nn * Ftm * 4);

  // Block 0: C=3, ct=72 padded to 128.  Block 1: C=64, ct=1536.
  run_block(stream, x0, 3, 128, P0, cheb, w, w.H0);
  run_block(stream, w.H0, 64, 1536, P1, cheb, w, w.XTAT);   // H1 aliases XTAT

  long long n = (long long)Bb * Nn * OUTD;
  k_final<<<gsz(n), 256, 0, stream>>>(w.XTAT, fw, fb, out, (int)n);
}
